// DL_JReg_85237920957128
// MI455X (gfx1250) — matrix operand order repaired and checked by reading, not compiled, not run on hardware
//
#include <hip/hip_runtime.h>
#include <hip/hip_bf16.h>
#include <stddef.h>
#include <stdint.h>

// ---- CDNA5 WMMA types -------------------------------------------------------
typedef __attribute__((ext_vector_type(16))) _Float16 v16h;
typedef __attribute__((ext_vector_type(8)))  _Float16 h8;
typedef __attribute__((ext_vector_type(8)))  float    v8f;

#define BATCH 2048
#define NVERT 6890
#define NJ    24
#define MB    128    // block rows (batch)
#define NB    128    // block cols (hidden dim)
#define KT    96     // k-tile: 32 vertices * 3 comps = 3 WMMA k-steps
#define VPT   32     // vertices per k-tile

struct GemmParams {
    const float* W1[NJ];
    const float* b1[NJ];
    const float* W2[NJ];
    const int*   idx[NJ];
    int          d[NJ];
};

struct MeanParams {
    const float* b2[NJ];
    const int*   idx[NJ];
    int          n[NJ];
};

// Pack two f32 -> two f16 in one 32-bit word.
static __device__ __forceinline__ uint32_t pk_h2(float a, float b) {
    union { _Float16 h[2]; uint32_t w; } u;
    u.h[0] = (_Float16)a;
    u.h[1] = (_Float16)b;
    return u.w;
}

// ---------------------------------------------------------------------------
// Kernel 1: per-(joint, batch) mean over selected vertices. One wave32 per
// (j,b) pair. Also seeds out[b][j][c] = mean + b2 so the GEMM kernel can
// accumulate with atomics.
// ---------------------------------------------------------------------------
__global__ __launch_bounds__(256) void jreg_mean_kernel(
    const float* __restrict__ verts, MeanParams P,
    float* __restrict__ means, float* __restrict__ out)
{
    const int t    = threadIdx.x;
    const int lane = t & 31;
    const int w    = (blockIdx.x << 3) + (t >> 5);   // global wave id, 0..B*J-1
    const int b    = w & (BATCH - 1);                // B is a power of two
    const int j    = w >> 11;
    if (j >= NJ) return;

    const int  n   = P.n[j];
    const int* idx = P.idx[j];

    float s0 = 0.f, s1 = 0.f, s2 = 0.f;
    for (int u = lane; u < n; u += 32) {
        const float* vp = &verts[((size_t)b * NVERT + (size_t)idx[u]) * 3];
        s0 += vp[0];
        s1 += vp[1];
        s2 += vp[2];
    }
    #pragma unroll
    for (int off = 16; off >= 1; off >>= 1) {
        s0 += __shfl_xor(s0, off, 32);
        s1 += __shfl_xor(s1, off, 32);
        s2 += __shfl_xor(s2, off, 32);
    }
    if (lane == 0) {
        const float inv = 1.0f / (float)n;
        const float m0 = s0 * inv, m1 = s1 * inv, m2 = s2 * inv;
        float* mp = &means[((size_t)j * BATCH + (size_t)b) * 3];
        mp[0] = m0; mp[1] = m1; mp[2] = m2;
        const float* b2 = P.b2[j];
        float* op = &out[((size_t)b * NJ + (size_t)j) * 3];
        op[0] = m0 + b2[0];
        op[1] = m1 + b2[1];
        op[2] = m2 + b2[2];
    }
}

// ---------------------------------------------------------------------------
// Kernel 2: fused   H = relu(X W1^T + b1) ;  out += H W2^T
// Block tile 128x128, 8 waves as 4x2 (wave tile 32 rows x 64 cols),
// 24 x v_wmma_f32_16x16x32_f16 per k-tile of 96 (= 32 vertices).
// X gather is per-vertex and branchless: clamp slot into [0,n), always load,
// select-to-zero past the end; 2 vertex slots x 8 rows per thread; centered
// f16 pairs written as aligned ds_store_b32.
// The 3-wide second GEMM folds into the epilogue (shuffle reduce + atomics).
// grid = (12, B/MB, NJ), block = 256.
// ---------------------------------------------------------------------------
__global__ __launch_bounds__(256) void jreg_gemm_kernel(
    const float* __restrict__ verts, const float* __restrict__ means,
    float* __restrict__ out, GemmParams P)
{
    const int j    = blockIdx.z;
    const int d    = P.d[j];
    const int dpad = (d + KT - 1) / KT * KT;
    const int n0   = blockIdx.x * NB;
    if (n0 >= dpad) return;                         // uniform per block
    const int b0   = blockIdx.y * MB;

    const float* __restrict__ W1  = P.W1[j];
    const float* __restrict__ b1v = P.b1[j];
    const float* __restrict__ W2  = P.W2[j];
    const int*   __restrict__ idx = P.idx[j];
    const int n = d / 3;

    __shared__ __align__(16) _Float16 Xs[MB * KT];   // 24 KB
    __shared__ __align__(16) _Float16 Ws[NB * KT];   // 24 KB
    __shared__ int idx_s[512];                       //  2 KB

    const int t = threadIdx.x;
    for (int u = t; u < n; u += 256) idx_s[u] = idx[u];

    const int lane = t & 31;
    const int wid  = t >> 5;
    const int wm   = wid & 3;      // wave row (0..3) -> 32 batch rows
    const int wn   = wid >> 2;     // wave col (0..1) -> 64 hidden cols
    const int lh   = lane >> 4;    // lane half
    const int l15  = lane & 15;

    // X staging: two adjacent vertex slots, 8 rows per thread.
    const int vpair = t & 15;             // vertex slots 2*vpair, 2*vpair+1
    const int xrow0 = t >> 4;             // rows xrow0 + 16u, u=0..7

    // Per-row means live in registers for the whole k-loop.
    float mr[8][3];
    #pragma unroll
    for (int u = 0; u < 8; ++u) {
        const int row = xrow0 + 16 * u;
        const float* mp = &means[((size_t)j * BATCH + (size_t)(b0 + row)) * 3];
        mr[u][0] = mp[0]; mr[u][1] = mp[1]; mr[u][2] = mp[2];
    }
    __syncthreads();   // idx_s ready

    v8f acc[2][4] = {};   // [m-subtile][n-subtile]

    const int ntiles = dpad / KT;
    for (int ti = 0; ti < ntiles; ++ti) {
        const int k0 = ti * KT;

        // ---- stage X tile: 2 idx lookups, 16 contiguous b96 loads ----
        {
            const int v0g = ti * VPT + 2 * vpair;
            const int v1g = v0g + 1;
            const int nm1 = n - 1;
            const int vi0 = idx_s[v0g < nm1 ? v0g : nm1];
            const int vi1 = idx_s[v1g < nm1 ? v1g : nm1];
            const float in0 = (v0g < n) ? 1.f : 0.f;   // select, not branch
            const float in1 = (v1g < n) ? 1.f : 0.f;
            #pragma unroll
            for (int u = 0; u < 8; ++u) {
                const int row = xrow0 + 16 * u;
                const size_t rbase = (size_t)(b0 + row) * NVERT;
                const float* p0 = &verts[(rbase + (size_t)vi0) * 3];
                const float* p1 = &verts[(rbase + (size_t)vi1) * 3];
                const float x0 = (p0[0] - mr[u][0]) * in0;
                const float y0 = (p0[1] - mr[u][1]) * in0;
                const float z0 = (p0[2] - mr[u][2]) * in0;
                const float x1 = (p1[0] - mr[u][0]) * in1;
                const float y1 = (p1[1] - mr[u][1]) * in1;
                const float z1 = (p1[2] - mr[u][2]) * in1;
                uint32_t* xp = (uint32_t*)&Xs[row * KT + vpair * 6];
                xp[0] = pk_h2(x0, y0);
                xp[1] = pk_h2(z0, x1);
                xp[2] = pk_h2(y1, z1);
            }
        }

        // ---- stage W1 tile: wave per row, lane covers 3 consecutive k ----
        {
            const int kk = lane * 3;
            #pragma unroll
            for (int pass = 0; pass < 16; ++pass) {
                const int row = wid + 8 * pass;       // 0..127
                const int i   = n0 + row;
                float w0 = 0.f, w1 = 0.f, w2 = 0.f;
                if (i < d) {
                    const float* wp = &W1[(size_t)i * d + (size_t)(k0 + kk)];
                    if (k0 + KT <= d) {               // fast path: full tile
                        w0 = wp[0]; w1 = wp[1]; w2 = wp[2];
                    } else {
                        if (k0 + kk + 0 < d) w0 = wp[0];
                        if (k0 + kk + 1 < d) w1 = wp[1];
                        if (k0 + kk + 2 < d) w2 = wp[2];
                    }
                }
                _Float16* sp = &Ws[row * KT + kk];
                sp[0] = (_Float16)w0;
                sp[1] = (_Float16)w1;
                sp[2] = (_Float16)w2;
            }
        }
        __syncthreads();

        // ---- 3 k-steps x (2x4) WMMA ----
        #pragma unroll
        for (int ks = 0; ks < 3; ++ks) {
            const int kb = ks * 32;
            v16h afrag[2], bfrag[4];
            #pragma unroll
            for (int mt = 0; mt < 2; ++mt) {
                const int m = wm * 32 + mt * 16 + l15;
                const h8 alo = *(const h8*)&Xs[m * KT + kb + lh * 16];
                const h8 ahi = *(const h8*)&Xs[m * KT + kb + lh * 16 + 8];
                #pragma unroll
                for (int u = 0; u < 8; ++u) {
                    afrag[mt][u] = alo[u]; afrag[mt][8 + u] = ahi[u];
                }
            }
            #pragma unroll
            for (int nt = 0; nt < 4; ++nt) {
                const int nn = wn * 64 + nt * 16 + l15;
                const h8 blo = *(const h8*)&Ws[nn * KT + kb + lh * 16];
                const h8 bhi = *(const h8*)&Ws[nn * KT + kb + lh * 16 + 8];
                #pragma unroll
                for (int u = 0; u < 8; ++u) {
                    bfrag[nt][u] = blo[u]; bfrag[nt][8 + u] = bhi[u];
                }
            }
            #pragma unroll
            for (int mt = 0; mt < 2; ++mt)
                #pragma unroll
                for (int nt = 0; nt < 4; ++nt)
                    acc[mt][nt] = __builtin_amdgcn_wmma_f32_16x16x32_f16(
                        false, afrag[mt], false, bfrag[nt],
                        (short)0, acc[mt][nt], false, false);
        }
        __syncthreads();
    }

    // ---- epilogue: h = relu(acc + b1); out[b][j][:] += sum_i h * W2[:, i]
    // C/D layout: VGPR e holds row e (lanes 0-15) and row e+8 (lanes 16-31),
    // col = lane%16. Branchless OOB (clamp + select), reduce over the 16-lane
    // column group, atomically add.
    const int i_base = n0 + wn * 64;
    #pragma unroll
    for (int mt = 0; mt < 2; ++mt) {
        #pragma unroll
        for (int e = 0; e < 8; ++e) {
            float s0 = 0.f, s1 = 0.f, s2 = 0.f;
            #pragma unroll
            for (int nt = 0; nt < 4; ++nt) {
                const int i  = i_base + nt * 16 + l15;
                const int ic = i < d ? i : d - 1;
                float h = acc[mt][nt][e] + b1v[ic];
                h = h > 0.f ? h : 0.f;
                h = (i < d) ? h : 0.f;
                s0 += h * W2[ic];
                s1 += h * W2[(size_t)d + ic];
                s2 += h * W2[2 * (size_t)d + ic];
            }
            #pragma unroll
            for (int off = 8; off >= 1; off >>= 1) {   // stays within 16-group
                s0 += __shfl_xor(s0, off, 32);
                s1 += __shfl_xor(s1, off, 32);
                s2 += __shfl_xor(s2, off, 32);
            }
            if (l15 == 0) {
                const int b = b0 + wm * 32 + mt * 16 + lh * 8 + e;
                float* o = &out[((size_t)b * NJ + (size_t)j) * 3];
                atomicAdd(o + 0, s0);
                atomicAdd(o + 1, s1);
                atomicAdd(o + 2, s2);
            }
        }
    }
}

// ---------------------------------------------------------------------------
// d_in layout (setup_inputs dict order, tuples flattened):
//   [0] vertices, [1..24] W1s, [25..48] b1s, [49..72] W2s, [73..96] b2s,
//   [97..120] joint_idx.  out: (B, J, 3) f32.  ws: >= B*J*3 floats (means).
// ---------------------------------------------------------------------------
extern "C" void kernel_launch(void* const* d_in, const int* in_sizes, int n_in,
                              void* d_out, int out_size, void* d_ws, size_t ws_size,
                              hipStream_t stream) {
    (void)n_in; (void)out_size; (void)ws_size;
    const float* verts = (const float*)d_in[0];

    GemmParams gp;
    MeanParams mp;
    for (int j = 0; j < NJ; ++j) {
        gp.W1[j]  = (const float*)d_in[1 + j];
        gp.b1[j]  = (const float*)d_in[25 + j];
        gp.W2[j]  = (const float*)d_in[49 + j];
        gp.idx[j] = (const int*)d_in[97 + j];
        const int n = in_sizes[97 + j];
        gp.d[j]   = 3 * n;
        mp.b2[j]  = (const float*)d_in[73 + j];
        mp.idx[j] = gp.idx[j];
        mp.n[j]   = n;
    }

    float* means = (float*)d_ws;                 // B*J*3 floats = ~590 KB
    float* out   = (float*)d_out;

    // One wave32 per (j,b): B*J = 49152 waves -> 6144 blocks of 8 waves.
    jreg_mean_kernel<<<dim3((BATCH * NJ) / 8), 256, 0, stream>>>(verts, mp, means, out);

    // Max d = 3*512 = 1536 -> 12 column tiles of 128; short joints early-exit.
    dim3 grid(12, BATCH / MB, NJ);
    jreg_gemm_kernel<<<grid, 256, 0, stream>>>(verts, means, out, gp);
}
